// CausalSelfAttention_35545149342070
// MI455X (gfx1250) — matrix operand order repaired and checked by reading, not compiled, not run on hardware
//
#include <hip/hip_runtime.h>
#include <hip/hip_bf16.h>
#include <math.h>

// Problem constants (B,S,D,H from reference)
#define BB 4
#define SS 2048
#define DD 1024
#define HH 16
#define HD 64

typedef __attribute__((ext_vector_type(16))) __bf16 v16bf;
typedef __attribute__((ext_vector_type(8)))  __bf16 v8bf;
typedef __attribute__((ext_vector_type(2)))  __bf16 v2bf;
typedef __attribute__((ext_vector_type(8)))  float  v8f;

#if defined(__has_builtin)
#if __has_builtin(__builtin_amdgcn_cvt_pk_bf16_f32)
#define HAVE_CVT_PK_BF16 1
#endif
#endif

__device__ __forceinline__ unsigned short f2bf(float f) {
#ifdef HAVE_CVT_PK_BF16
  v2bf p = __builtin_amdgcn_cvt_pk_bf16_f32(f, 0.f);
  return (unsigned short)(__builtin_bit_cast(unsigned, p) & 0xFFFFu);
#else
  union { float f; unsigned u; } v; v.f = f;
  unsigned u = v.u;
  unsigned r = u + 0x7FFFu + ((u >> 16) & 1u);      // round-to-nearest-even
  if ((u & 0x7F800000u) == 0x7F800000u) r = u;      // Inf/NaN passthrough
  return (unsigned short)(r >> 16);
#endif
}

// 16-lane row reductions via DPP row_ror (no LDS traffic, unlike ds_bpermute)
template <int CTRL>
__device__ __forceinline__ float dpp_rot(float x) {
  return __builtin_bit_cast(float,
      __builtin_amdgcn_update_dpp(0, __builtin_bit_cast(int, x), CTRL, 0xF, 0xF, true));
}
__device__ __forceinline__ float red_max16(float x) {
  x = fmaxf(x, dpp_rot<0x121>(x));   // row_ror:1
  x = fmaxf(x, dpp_rot<0x122>(x));   // row_ror:2
  x = fmaxf(x, dpp_rot<0x124>(x));   // row_ror:4
  x = fmaxf(x, dpp_rot<0x128>(x));   // row_ror:8
  return x;
}
__device__ __forceinline__ float red_add16(float x) {
  x += dpp_rot<0x121>(x);
  x += dpp_rot<0x122>(x);
  x += dpp_rot<0x124>(x);
  x += dpp_rot<0x128>(x);
  return x;
}

// Build a 16-wide bf16 A/B fragment from two contiguous 8-element runs.
__device__ __forceinline__ v16bf ld_frag(const unsigned short* p0, const unsigned short* p1) {
  v8bf lo = *(const v8bf*)p0;
  v8bf hi = *(const v8bf*)p1;
  return __builtin_shufflevector(lo, hi, 0,1,2,3,4,5,6,7,8,9,10,11,12,13,14,15);
}

__device__ __forceinline__ v8f wmma_bf16(v16bf a, v16bf b, v8f c) {
  return __builtin_amdgcn_wmma_f32_16x16x32_bf16(false, a, false, b, (short)0, c, false, false);
}

// ---------------- fp32 -> bf16 convert (vectorized) ----------------
__global__ __launch_bounds__(256) void cvt_f32_bf16(const float* __restrict__ in,
                                                    unsigned short* __restrict__ out, int n4) {
  int i = blockIdx.x * blockDim.x + threadIdx.x;
  if (i < n4) {
    float4 v = ((const float4*)in)[i];
    ushort4 o;
    o.x = f2bf(v.x); o.y = f2bf(v.y); o.z = f2bf(v.z); o.w = f2bf(v.w);
    ((ushort4*)out)[i] = o;
  }
}

// ---------------- W[k][n] fp32 -> Wt[n][k] bf16 (LDS tiled) ----------------
__global__ __launch_bounds__(256) void transpose_cvt(const float* __restrict__ W,
                                                     unsigned short* __restrict__ Wt) {
  __shared__ float tile[32][33];
  int x = blockIdx.x * 32 + threadIdx.x;  // n
  int y = blockIdx.y * 32 + threadIdx.y;  // k
#pragma unroll
  for (int j = 0; j < 32; j += 8)
    tile[threadIdx.y + j][threadIdx.x] = W[(size_t)(y + j) * DD + x];
  __syncthreads();
  x = blockIdx.y * 32 + threadIdx.x;  // k
  y = blockIdx.x * 32 + threadIdx.y;  // n
#pragma unroll
  for (int j = 0; j < 32; j += 8)
    Wt[(size_t)(y + j) * DD + x] = f2bf(tile[threadIdx.x][threadIdx.y + j]);
}

// ---------------- bf16 GEMM: C[M][N] = A[M][K] * Bt[N][K]^T ----------------
// MODE 0: store bf16 row-major         (Q, K projections)
// MODE 1: store bf16 V-transposed      Vt[(b*DD + n)][s]   (V projection)
// MODE 2: store fp32 + bias            (output projection -> d_out)
template <int MODE>
__global__ __launch_bounds__(128) void gemm_bf16(
    const unsigned short* __restrict__ A, const unsigned short* __restrict__ Bt,
    void* __restrict__ Out, const float* __restrict__ bias, int N, int K) {
  const int lane = threadIdx.x & 31;
  const int wave = threadIdx.x >> 5;
  const int c    = lane & 15;
  const int half = lane >> 4;
  const int m0 = (blockIdx.y * 4 + wave) * 16;
  const int n0 = blockIdx.x * 64;

  const v8f vzero = {0.f,0.f,0.f,0.f,0.f,0.f,0.f,0.f};
  v8f acc[4] = {vzero, vzero, vzero, vzero};

  // A-fragment (16x32 bf16): lanes 0-15 row m, K=k0..k0+15;
  // lanes 16-31 same rows, K=k0+16..k0+31.
  const unsigned short* arow = A + (size_t)(m0 + c) * K + half * 16;
  for (int k0 = 0; k0 < K; k0 += 32) {
    v16bf af = ld_frag(arow + k0, arow + k0 + 8);
#pragma unroll
    for (int t = 0; t < 4; ++t) {
      // B-fragment (32x16): lane holds column n = c, K-range selected by half.
      const unsigned short* bp = Bt + (size_t)(n0 + t * 16 + c) * K + k0 + half * 16;
      v16bf bf = *(const v16bf*)bp;
      acc[t] = wmma_bf16(af, bf, acc[t]);
    }
  }
  // C/D layout: VGPR r -> row m0 + half*8 + r, col = n0 + t*16 + c
#pragma unroll
  for (int t = 0; t < 4; ++t) {
#pragma unroll
    for (int r = 0; r < 8; ++r) {
      int row = m0 + half * 8 + r;
      int col = n0 + t * 16 + c;
      float v = acc[t][r];
      if (MODE == 0) {
        ((unsigned short*)Out)[(size_t)row * N + col] = f2bf(v);
      } else if (MODE == 1) {
        int b = row >> 11;            // row / S
        int s = row & (SS - 1);       // row % S
        ((unsigned short*)Out)[((size_t)b * DD + col) * SS + s] = f2bf(v);
      } else {
        ((float*)Out)[(size_t)row * N + col] = v + bias[col];
      }
    }
  }
}

// ---------------- flash attention (causal), bf16 WMMA ----------------
// grid (S/64, B*H), block 128: wave w owns queries [q0, q0+16) of head (b,h).
__global__ __launch_bounds__(128) void attn_fwd(
    const unsigned short* __restrict__ Q,   // [B*S][D] bf16
    const unsigned short* __restrict__ Kb,  // [B*S][D] bf16
    const unsigned short* __restrict__ Vt,  // [B*D][S] bf16 (per-head transposed)
    unsigned short* __restrict__ Ctx) {     // [B*S][D] bf16
  __shared__ __align__(16) unsigned short pbuf[4][16 * 32];
  const int lane = threadIdx.x & 31;
  const int wave = threadIdx.x >> 5;
  const int c    = lane & 15;
  const int half = lane >> 4;
  const int q0 = blockIdx.x * 64 + wave * 16;
  const int b  = blockIdx.y >> 4;   // / H
  const int h  = blockIdx.y & 15;   // % H

  // scores computed in base-2 domain: scale = HD^-0.5 * log2(e)
  const float SC = 0.125f * 1.44269504088896340736f;
  const float NEG_INF = -__builtin_inff();

  // Q fragments for d=0..31 and d=32..63
  const unsigned short* qrow = Q + (size_t)(b * SS + q0 + c) * DD + h * HD + half * 16;
  v16bf qf0 = ld_frag(qrow,      qrow + 8);
  v16bf qf1 = ld_frag(qrow + 32, qrow + 40);

  const v8f vzero = {0.f,0.f,0.f,0.f,0.f,0.f,0.f,0.f};
  v8f oacc[4] = {vzero, vzero, vzero, vzero};
  float mrow[8], lrow[8];
#pragma unroll
  for (int r = 0; r < 8; ++r) { mrow[r] = NEG_INF; lrow[r] = 0.f; }

  const int ntiles = (q0 + 47) >> 5;   // causal: only key tiles with keys <= q0+15
  for (int t = 0; t < ntiles; ++t) {
    const int kb = t * 32;
    const bool diag = (kb + 31 > q0);  // wave-uniform: does this tile touch the mask?
    // scores: S = Q @ K^T.  B-fragment column n = key, elements = d slice.
    v8f s0 = vzero, s1 = vzero;
    {
      const unsigned short* kr0 = Kb + (size_t)(b * SS + kb + c) * DD + h * HD + half * 16;
      const unsigned short* kr1 = kr0 + (size_t)16 * DD;   // keys kb+16+c
      s0 = wmma_bf16(qf0, *(const v16bf*)(kr0),      s0);
      s0 = wmma_bf16(qf1, *(const v16bf*)(kr0 + 32), s0);
      s1 = wmma_bf16(qf0, *(const v16bf*)(kr1),      s1);
      s1 = wmma_bf16(qf1, *(const v16bf*)(kr1 + 32), s1);
    }
    // online softmax; score row m lives in one VGPR across 16 lanes of a half-wave
#pragma unroll
    for (int r = 0; r < 8; ++r) {
      const int q = q0 + half * 8 + r;
      float a0 = s0[r] * SC;
      float a1 = s1[r] * SC;
      if (diag) {
        if (kb + c > q)      a0 = NEG_INF;
        if (kb + 16 + c > q) a1 = NEG_INF;
      }
      float rm = red_max16(fmaxf(a0, a1));
      float mnew = fmaxf(mrow[r], rm);
      float corr = __builtin_amdgcn_exp2f(mrow[r] - mnew);
      float p0 = __builtin_amdgcn_exp2f(a0 - mnew);
      float p1 = __builtin_amdgcn_exp2f(a1 - mnew);
      float rs = red_add16(p0 + p1);
      lrow[r] = lrow[r] * corr + rs;
      mrow[r] = mnew;
      oacc[0][r] *= corr; oacc[1][r] *= corr; oacc[2][r] *= corr; oacc[3][r] *= corr;
      // stage P tile (16x32 bf16) in per-wave LDS buffer, plain row-major
      pbuf[wave][(half * 8 + r) * 32 + c]      = f2bf(p0);
      pbuf[wave][(half * 8 + r) * 32 + 16 + c] = f2bf(p1);
    }
    // DS ops are in-order per wave; wait for stores before cross-lane reload
    asm volatile("s_wait_dscnt 0" ::: "memory");
    const unsigned short* pp = &pbuf[wave][c * 32 + half * 16];  // A-layout read
    v16bf pf = ld_frag(pp, pp + 8);
    // O += P @ V.  B-fragment column n = d, elements = key slice (contiguous in Vt)
#pragma unroll
    for (int dt = 0; dt < 4; ++dt) {
      const unsigned short* vp =
          Vt + (size_t)(b * DD + h * HD + dt * 16 + c) * SS + kb + half * 16;
      oacc[dt] = wmma_bf16(pf, *(const v16bf*)vp, oacc[dt]);
    }
  }
  // normalize and write context (row-major bf16)
#pragma unroll
  for (int dt = 0; dt < 4; ++dt) {
#pragma unroll
    for (int r = 0; r < 8; ++r) {
      float v = oacc[dt][r] / lrow[r];
      Ctx[(size_t)(b * SS + q0 + half * 8 + r) * DD + h * HD + dt * 16 + c] = f2bf(v);
    }
  }
}

// ---------------- host-side launch ----------------
extern "C" void kernel_launch(void* const* d_in, const int* in_sizes, int n_in,
                              void* d_out, int out_size, void* d_ws, size_t ws_size,
                              hipStream_t stream) {
  const float* x  = (const float*)d_in[0];
  const float* Wq = (const float*)d_in[1];
  const float* Wk = (const float*)d_in[2];
  const float* Wv = (const float*)d_in[3];
  const float* Wo = (const float*)d_in[4];
  const float* bo = (const float*)d_in[5];
  float* out = (float*)d_out;

  char* ws = (char*)d_ws;
  const size_t szX = (size_t)BB * SS * DD * sizeof(unsigned short);  // 16 MB
  const size_t szW = (size_t)DD * DD * sizeof(unsigned short);       // 2 MB
  unsigned short* xb  = (unsigned short*)ws; ws += szX;
  unsigned short* wqt = (unsigned short*)ws; ws += szW;
  unsigned short* wkt = (unsigned short*)ws; ws += szW;
  unsigned short* wvt = (unsigned short*)ws; ws += szW;
  unsigned short* wot = (unsigned short*)ws; ws += szW;
  unsigned short* qb  = (unsigned short*)ws; ws += szX;
  unsigned short* kb  = (unsigned short*)ws; ws += szX;
  unsigned short* vt  = (unsigned short*)ws; ws += szX;
  unsigned short* ctx = xb;  // x no longer needed after the QKV projections

  // 1) stage inputs as bf16
  int n4 = BB * SS * DD / 4;
  cvt_f32_bf16<<<(n4 + 255) / 256, 256, 0, stream>>>(x, xb, n4);
  dim3 tb(32, 8), tg(DD / 32, DD / 32);
  transpose_cvt<<<tg, tb, 0, stream>>>(Wq, wqt);
  transpose_cvt<<<tg, tb, 0, stream>>>(Wk, wkt);
  transpose_cvt<<<tg, tb, 0, stream>>>(Wv, wvt);
  transpose_cvt<<<tg, tb, 0, stream>>>(Wo, wot);

  // 2) QKV projections (WMMA bf16)
  dim3 gb(DD / 64, (BB * SS) / 64);  // (16, 128), 4 waves/block
  gemm_bf16<0><<<gb, 128, 0, stream>>>(xb, wqt, qb, nullptr, DD, DD);
  gemm_bf16<0><<<gb, 128, 0, stream>>>(xb, wkt, kb, nullptr, DD, DD);
  gemm_bf16<1><<<gb, 128, 0, stream>>>(xb, wvt, vt, nullptr, DD, DD);

  // 3) causal flash attention
  attn_fwd<<<dim3(SS / 64, BB * HH), 128, 0, stream>>>(qb, kb, vt, ctx);

  // 4) output projection + bias -> fp32 d_out
  gemm_bf16<2><<<gb, 128, 0, stream>>>(ctx, wot, out, bo, DD, DD);
}
